// RelationalTransformer_24008867184815
// MI455X (gfx1250) — compile-verified, matrix operand order checked by reading, not run on hardware
//
#include <hip/hip_runtime.h>
#include <math.h>

// ---------------------------------------------------------------------------
// RelationalTransformer forward loss for MI455X (gfx1250, wave32).
// GEMM tiles staged to LDS by the Tensor Data Mover (tensor_load_to_lds),
// compute via v_wmma_f32_16x16x32_f16 (f16 in, f32 accumulate).
// ---------------------------------------------------------------------------

typedef __attribute__((ext_vector_type(16))) _Float16 v16h;
typedef __attribute__((ext_vector_type(8)))  float    v8f;
typedef __attribute__((ext_vector_type(4)))  int      v4i;
typedef __attribute__((ext_vector_type(8)))  int      v8i;
typedef unsigned int       u32;
typedef unsigned long long u64;

#define B_    4
#define S_    1024
#define M_    (B_*S_)        // 4096 tokens
#define DM    256
#define NH    8
#define DH    32
#define DFF   1024
#define DT    384
#define KNBR  8
#define EPSV  1.1920929e-07f

// ---------------------------------------------------------------------------
// Block-wide sum over 256 threads (8 waves of 32).
// ---------------------------------------------------------------------------
__device__ __forceinline__ float block_sum_256(float v) {
    __shared__ float red[8];
    #pragma unroll
    for (int off = 16; off; off >>= 1) v += __shfl_xor(v, off, 32);
    const int wid = threadIdx.x >> 5;
    if ((threadIdx.x & 31) == 0) red[wid] = v;
    __syncthreads();
    float t = 0.f;
    #pragma unroll
    for (int i = 0; i < 8; ++i) t += red[i];
    __syncthreads();
    return t;
}

// ---------------------------------------------------------------------------
// TDM: DMA a 2D f32 tile (rows x 32) from global to LDS with 1-DWORD padding
// after every 32 DWORDs (row stride in LDS = 33 DWORDs, odd => no bank
// conflicts on fragment gathers). Descriptor per CDNA5 ISA 8.3/8.4:
//   group0: count=1, lds_addr, global_addr[56:0], type=2
//   group1: data_size=4B, pad_enable, pad_interval=32DW, pad_amount=1DW,
//           tensor_dim0=32, tensor_dim1=rows, tile_dim0=32, tile_dim1=rows,
//           tensor_dim0_stride=K elements
// Issued by one wave; completion via s_wait_tensorcnt + workgroup barrier.
// ---------------------------------------------------------------------------
__device__ __forceinline__ void tdm_load_tile32(u32 lds_byte_addr, u64 gaddr,
                                                u32 stride_elems, u32 rows)
{
    v4i g0, gz;
    v8i g1;
    gz[0] = 0; gz[1] = 0; gz[2] = 0; gz[3] = 0;
    g0[0] = 1;                                            // count=1, no gather
    g0[1] = (int)lds_byte_addr;                           // lds_addr
    g0[2] = (int)(u32)gaddr;                              // global_addr[31:0]
    g0[3] = (int)(((u32)(gaddr >> 32) & 0x01FFFFFFu)      // global_addr[56:32]
                  | 0x80000000u);                         // type=2 (image)
    g1[0] = (2 << 16) | (1 << 20) | (4 << 22);            // 4B, pad 1DW/32DW
    g1[1] = (int)(32u << 16);                             // tensor_dim0 = 32
    g1[2] = (int)((rows & 0xFFFFu) << 16);                // tensor_dim1 lo16
    g1[3] = (int)(((rows >> 16) & 0xFFFFu) | (32u << 16));// tile_dim0 = 32
    g1[4] = (int)(rows & 0xFFFFu);                        // tile_dim1, tile_dim2=0
    g1[5] = (int)stride_elems;                            // dim0_stride lo32
    g1[6] = 0;                                            // stride hi, dim1_stride
    g1[7] = 0;
    asm volatile("tensor_load_to_lds %0, %1, %2, %3"
                 :: "s"(g0), "s"(g1), "s"(gz), "s"(gz) : "memory");
}

// ---------------------------------------------------------------------------
// WMMA GEMM:  C[M,N] = A[M,K] @ W[N,K]^T  (+ bias[col]) (+= if accum)
// Workgroup: 256 threads = 8 waves, tile 256(M) x 64(N), K step 32.
// Each wave owns a 32-row stripe: 2 A-frags x 4 B-frags = 8 wmma / K-step.
// Fragment layouts per CDNA5 ISA 7.12.2 (16-bit A 16x32; B mirrored).
// ---------------------------------------------------------------------------
__global__ __launch_bounds__(256)
void k_gemm(const float* __restrict__ A, const float* __restrict__ W,
            const float* __restrict__ bias, float* __restrict__ C,
            int M, int N, int K, int accum)
{
    __shared__ float sAf[256 * 33];     // TDM-padded: 33 DWORDs per 32-elem row
    __shared__ float sBf[64 * 33];

    const int tid  = threadIdx.x;
    const int wid  = tid >> 5;
    const int lane = tid & 31;
    const int half = lane >> 4;
    const int ml   = lane & 15;
    const int m0   = blockIdx.y * 256;
    const int n0   = blockIdx.x * 64;

    v8f acc[2][4];
    #pragma unroll
    for (int mt = 0; mt < 2; ++mt)
        #pragma unroll
        for (int nt = 0; nt < 4; ++nt)
            #pragma unroll
            for (int r = 0; r < 8; ++r) acc[mt][nt][r] = 0.f;

    for (int kt = 0; kt < K; kt += 32) {
        if (wid == 0) {
            tdm_load_tile32((u32)(uintptr_t)sAf,
                            (u64)(uintptr_t)(A + (size_t)m0 * K + kt),
                            (u32)K, 256u);
            tdm_load_tile32((u32)(uintptr_t)sBf,
                            (u64)(uintptr_t)(W + (size_t)n0 * K + kt),
                            (u32)K, 64u);
            __builtin_amdgcn_s_wait_tensorcnt(0);
        }
        __syncthreads();                 // publish LDS tiles to all waves

        // ---- A fragments (2 row-tiles per wave): lane = M row, packed K
        v16h af[2];
        #pragma unroll
        for (int mt = 0; mt < 2; ++mt) {
            const int rowb = (wid * 32 + mt * 16 + ml) * 33;
            #pragma unroll
            for (int e = 0; e < 16; ++e) {
                int i = e >> 1, s = e & 1;
                int kk = ((i >> 2) << 4) + ((i & 3) << 1) + (half << 3) + s;
                af[mt][e] = (_Float16)sAf[rowb + kk];
            }
        }
        // ---- 4 N-tiles, each B fragment feeds both A fragments
        #pragma unroll
        for (int nt = 0; nt < 4; ++nt) {
            v16h bf;
            const int rowb = (nt * 16 + ml) * 33;
            #pragma unroll
            for (int e = 0; e < 16; ++e) {
                int i = e >> 1, s = e & 1;
                int kk = ((i >> 2) << 4) + ((i & 3) << 1) + (half << 3) + s;
                bf[e] = (_Float16)sBf[rowb + kk];
            }
            acc[0][nt] = __builtin_amdgcn_wmma_f32_16x16x32_f16(
                false, af[0], false, bf, (short)0, acc[0][nt], false, false);
            acc[1][nt] = __builtin_amdgcn_wmma_f32_16x16x32_f16(
                false, af[1], false, bf, (short)0, acc[1][nt], false, false);
        }
        __syncthreads();                 // drain readers before next TDM write
    }

    // ---- epilogue: C/D layout: vgpr r -> row r (lanes 0-15) / r+8 (16-31)
    #pragma unroll
    for (int mt = 0; mt < 2; ++mt) {
        #pragma unroll
        for (int nt = 0; nt < 4; ++nt) {
            #pragma unroll
            for (int r = 0; r < 8; ++r) {
                int row = m0 + wid * 32 + mt * 16 + r + half * 8;
                int col = n0 + nt * 16 + ml;
                float v = acc[mt][nt][r];
                if (bias) v += bias[col];
                size_t idx = (size_t)row * N + col;
                if (accum) C[idx] += v; else C[idx] = v;
            }
        }
    }
}

// ---------------------------------------------------------------------------
// RMSNorm: one 256-thread block per token row of width DM=256.
// ---------------------------------------------------------------------------
__global__ __launch_bounds__(256)
void k_rmsnorm(const float* __restrict__ X, const float* __restrict__ w,
               float* __restrict__ O)
{
    const int row = blockIdx.x, d = threadIdx.x;
    float v  = X[(size_t)row * DM + d];
    float ss = block_sum_256(v * v);
    float r  = rsqrtf(ss / DM + EPSV);
    O[(size_t)row * DM + d] = v * r * w[d];
}

// Encoder init: x = rmsnorm(tmp, norm) * notpad
__global__ __launch_bounds__(256)
void k_enc_colname(const float* __restrict__ T, const float* __restrict__ norm,
                   const unsigned char* __restrict__ ispad, float* __restrict__ x)
{
    const int row = blockIdx.x, d = threadIdx.x;
    float e  = T[(size_t)row * DM + d];
    float ss = block_sum_256(e * e);
    float r  = rsqrtf(ss / DM + EPSV);
    float np = ispad[row] ? 0.f : 1.f;
    x[(size_t)row * DM + d] = e * r * norm[d] * np;
}

// Scalar-type encoder (T_IN==1): x += rms(v*w+b)*sel&~m + memb*sel&m
__global__ __launch_bounds__(256)
void k_enc_scalar(const float* __restrict__ vals, const float* __restrict__ w,
                  const float* __restrict__ b, const float* __restrict__ norm,
                  const float* __restrict__ memb, const int* __restrict__ sem,
                  const unsigned char* __restrict__ ispad,
                  const unsigned char* __restrict__ mks, int code,
                  float* __restrict__ x)
{
    const int row = blockIdx.x, d = threadIdx.x;
    float e  = vals[row] * w[d] + b[d];
    float ss = block_sum_256(e * e);
    float r  = rsqrtf(ss / DM + EPSV);
    float en = e * r * norm[d];
    bool sel = (sem[row] == code) && !ispad[row];
    bool mk  = mks[row] != 0;
    float add = sel ? (mk ? memb[d] : en) : 0.f;
    x[(size_t)row * DM + d] += add;
}

// Vector-type encoder (text): encoded vector precomputed by GEMM
__global__ __launch_bounds__(256)
void k_enc_vec(const float* __restrict__ T, const float* __restrict__ norm,
               const float* __restrict__ memb, const int* __restrict__ sem,
               const unsigned char* __restrict__ ispad,
               const unsigned char* __restrict__ mks, int code,
               float* __restrict__ x)
{
    const int row = blockIdx.x, d = threadIdx.x;
    float e  = T[(size_t)row * DM + d];
    float ss = block_sum_256(e * e);
    float r  = rsqrtf(ss / DM + EPSV);
    float en = e * r * norm[d];
    bool sel = (sem[row] == code) && !ispad[row];
    bool mk  = mks[row] != 0;
    float add = sel ? (mk ? memb[d] : en) : 0.f;
    x[(size_t)row * DM + d] += add;
}

// ---------------------------------------------------------------------------
// Flash attention with on-the-fly relational masks.
// One wave per (b, h, q) row; lanes stripe keys; online softmax; butterfly
// merge across the 32 lanes. layer: 0=col 1=feat 2=nbr 3=full.
// ---------------------------------------------------------------------------
__global__ __launch_bounds__(256)
void k_attn(const float* __restrict__ Q, const float* __restrict__ Kb,
            const float* __restrict__ V, const int* __restrict__ node,
            const int* __restrict__ f2p, const int* __restrict__ colid,
            const int* __restrict__ tblid,
            const unsigned char* __restrict__ ispad,
            int layer, float* __restrict__ O)
{
    const int wid  = threadIdx.x >> 5;
    const int lane = threadIdx.x & 31;
    const int row  = blockIdx.x * 8 + wid;          // (b*NH + h)*S + q
    const int q    = row & (S_ - 1);
    const int h    = (row >> 10) & (NH - 1);
    const int b    = row >> 13;
    const int bS   = b * S_;

    float qv[DH];
    const float* qp = Q + (size_t)(bS + q) * DM + h * DH;
    #pragma unroll
    for (int d = 0; d < DH; ++d) qv[d] = qp[d];

    const int  nq   = node[bS + q];
    const int  cq   = colid[bS + q];
    const int  tq   = tblid[bS + q];
    const bool padq = (ispad[bS + q] == 0);
    int f2pq[KNBR];
    #pragma unroll
    for (int j = 0; j < KNBR; ++j) f2pq[j] = f2p[(bS + q) * KNBR + j];

    float mrun = -INFINITY, lrun = 0.f;
    float ov[DH];
    #pragma unroll
    for (int d = 0; d < DH; ++d) ov[d] = 0.f;
    const float scale = 0.17677669529663687f;       // 1/sqrt(32)

    for (int key = lane; key < S_; key += 32) {
        bool pm = padq && (ispad[bS + key] == 0);
        bool msk;
        if (layer == 0) {
            msk = pm && (colid[bS + key] == cq) && (tblid[bS + key] == tq);
        } else if (layer == 1) {
            int nk = node[bS + key];
            bool any = (nk == nq);
            #pragma unroll
            for (int j = 0; j < KNBR; ++j) any = any || (f2pq[j] == nk);
            msk = pm && any;
        } else if (layer == 2) {
            bool any = false;
            #pragma unroll
            for (int j = 0; j < KNBR; ++j)
                any = any || (f2p[(bS + key) * KNBR + j] == nq);
            msk = pm && any;
        } else {
            msk = pm;
        }
        if (msk) {
            const float* kp = Kb + (size_t)(bS + key) * DM + h * DH;
            float s = 0.f;
            #pragma unroll
            for (int d = 0; d < DH; ++d) s += qv[d] * kp[d];
            s *= scale;
            float mn   = fmaxf(mrun, s);
            float corr = __expf(mrun - mn);   // exp(-inf)=0 on first hit
            float p    = __expf(s - mn);
            const float* vp = V + (size_t)(bS + key) * DM + h * DH;
            #pragma unroll
            for (int d = 0; d < DH; ++d) ov[d] = ov[d] * corr + p * vp[d];
            lrun = lrun * corr + p;
            mrun = mn;
        }
    }

    // merge lane-local flash states across the wave
    #pragma unroll
    for (int off = 16; off; off >>= 1) {
        float m2 = __shfl_xor(mrun, off, 32);
        float l2 = __shfl_xor(lrun, off, 32);
        float mn = fmaxf(mrun, m2);
        bool  dead = (mrun == -INFINITY) && (m2 == -INFINITY);
        float c1 = dead ? 0.f : __expf(mrun - mn);
        float c2 = dead ? 0.f : __expf(m2 - mn);
        #pragma unroll
        for (int d = 0; d < DH; ++d) {
            float o2 = __shfl_xor(ov[d], off, 32);
            ov[d] = ov[d] * c1 + o2 * c2;
        }
        lrun = lrun * c1 + l2 * c2;
        mrun = mn;
    }

    if (lane == 0) {
        float inv = (lrun > 0.f) ? 1.f / lrun : 0.f;   // all-masked row -> 0
        float* op = O + (size_t)(bS + q) * DM + h * DH;
        #pragma unroll
        for (int d = 0; d < DH; ++d) op[d] = ov[d] * inv;
    }
}

// SwiGLU gate: h1 = silu(h1) * h3
__global__ void k_silu_mul(float* __restrict__ h1, const float* __restrict__ h3,
                           int n)
{
    int i = blockIdx.x * blockDim.x + threadIdx.x;
    if (i < n) {
        float a = h1[i];
        h1[i] = (a / (1.f + __expf(-a))) * h3[i];
    }
}

// Decoder heads + Huber / BCE loss, accumulated into scal[0]; count in scal[1]
__global__ __launch_bounds__(256)
void k_loss(const float* __restrict__ XN,
            const float* __restrict__ wn, const float* __restrict__ bn,
            const float* __restrict__ wd, const float* __restrict__ bd,
            const float* __restrict__ wb, const float* __restrict__ bb,
            const float* __restrict__ numv, const float* __restrict__ dtv,
            const float* __restrict__ boolv, const int* __restrict__ sem,
            const unsigned char* __restrict__ mks, float* __restrict__ scal)
{
    const int row = blockIdx.x, d = threadIdx.x;
    float xv = XN[(size_t)row * DM + d];
    float t0 = block_sum_256(xv * wn[d]);
    float t1 = block_sum_256(xv * wd[d]);
    float t2 = block_sum_256(xv * wb[d]);
    if (threadIdx.x == 0) {
        bool mk = mks[row] != 0;
        if (mk) {
            int st = sem[row];
            float lt = 0.f;
            if (st == 0) {                       // number: Huber
                float dd = (t0 + bn[0]) - numv[row];
                float a = fabsf(dd);
                lt = (a < 1.f) ? 0.5f * dd * dd : (a - 0.5f);
            } else if (st == 2) {                // datetime: Huber
                float dd = (t1 + bd[0]) - dtv[row];
                float a = fabsf(dd);
                lt = (a < 1.f) ? 0.5f * dd * dd : (a - 0.5f);
            } else if (st == 3) {                // boolean: BCE-with-logits
                float yh  = t2 + bb[0];
                float tgt = (boolv[row] > 0.f) ? 1.f : 0.f;
                lt = fmaxf(yh, 0.f) - yh * tgt + log1pf(__expf(-fabsf(yh)));
            }
            atomicAdd(&scal[0], lt);
            atomicAdd(&scal[1], 1.0f);
        }
    }
}

__global__ void k_final(const float* __restrict__ scal, float* __restrict__ out)
{
    out[0] = scal[0] / scal[1];
}

// ---------------------------------------------------------------------------
// Host-side orchestration
// ---------------------------------------------------------------------------
static inline void gemm(const float* A, const float* W, const float* bias,
                        float* C, int M, int N, int K, int accum,
                        hipStream_t s)
{
    dim3 g(N / 64, M / 256);
    k_gemm<<<g, 256, 0, s>>>(A, W, bias, C, M, N, K, accum);
}

extern "C" void kernel_launch(void* const* d_in, const int* in_sizes, int n_in,
                              void* d_out, int out_size, void* d_ws, size_t ws_size,
                              hipStream_t stream)
{
    (void)in_sizes; (void)n_in; (void)out_size; (void)ws_size;

    // ---- inputs (setup_inputs dict order; params pytree with sorted keys)
    const float* col_name_values = (const float*)d_in[0];
    const float* number_values   = (const float*)d_in[1];
    const float* text_values     = (const float*)d_in[2];
    const float* datetime_values = (const float*)d_in[3];
    const float* boolean_values  = (const float*)d_in[4];
    const int P = 5;                                  // params leaves start
    auto F = [&](int i) { return (const float*)d_in[P + i]; };
    // params leaf map (sorted-key pytree):
    //  per block (24 leaves, block b at base=24*b):
    //    attns: col{wk,wo,wq,wv}=0..3 feat=4..7 full=8..11 nbr=12..15
    //    ffn: w1=16 w2=17 w3=18
    //    norms: col=19 feat=20 ffn=21 full=22 nbr=23
    //  48..55: dec {boolean,datetime,number,text} x {b,w}
    //  56..65: enc {boolean,col_name,datetime,number,text} x {b,w}
    //  66..70: enc_norm {boolean,col_name,datetime,number,text}
    //  71..74: mask_emb {boolean,datetime,number,text}
    //  75:     norm_out
    const int*           node  = (const int*)d_in[P + 76];
    const int*           f2p   = (const int*)d_in[P + 77];
    const int*           colid = (const int*)d_in[P + 78];
    const int*           tblid = (const int*)d_in[P + 79];
    const unsigned char* ispad = (const unsigned char*)d_in[P + 80];
    const int*           sem   = (const int*)d_in[P + 81];
    const unsigned char* mks   = (const unsigned char*)d_in[P + 82];

    // ---- workspace layout (4 MB units for [4096,256] f32)
    const size_t XB = (size_t)M_ * DM * sizeof(float);    // 4 MiB
    char* w0 = (char*)d_ws;
    float* x   = (float*)(w0 + 0 * XB);
    float* xn  = (float*)(w0 + 1 * XB);
    float* qb  = (float*)(w0 + 2 * XB);
    float* kb  = (float*)(w0 + 3 * XB);
    float* vb  = (float*)(w0 + 4 * XB);
    float* ao  = (float*)(w0 + 5 * XB);
    float* h1  = (float*)(w0 + 6 * XB);                   // 16 MiB
    float* h3  = (float*)(w0 + 10 * XB);                  // 16 MiB
    float* scal = (float*)(w0 + 14 * XB);                 // 2 floats

    hipMemsetAsync(scal, 0, 2 * sizeof(float), stream);

    // ---- encode ----------------------------------------------------------
    gemm(col_name_values, F(59), F(58), qb, M_, DM, DT, 0, stream); // col_name
    k_enc_colname<<<M_, 256, 0, stream>>>(qb, F(67), ispad, x);
    k_enc_scalar<<<M_, 256, 0, stream>>>(number_values,   F(63), F(62), F(69),
                                         F(73), sem, ispad, mks, 0, x);
    gemm(text_values, F(65), F(64), qb, M_, DM, DT, 0, stream);     // text
    k_enc_vec<<<M_, 256, 0, stream>>>(qb, F(70), F(74), sem, ispad, mks, 1, x);
    k_enc_scalar<<<M_, 256, 0, stream>>>(datetime_values, F(61), F(60), F(68),
                                         F(72), sem, ispad, mks, 2, x);
    k_enc_scalar<<<M_, 256, 0, stream>>>(boolean_values,  F(57), F(56), F(66),
                                         F(71), sem, ispad, mks, 3, x);

    // ---- transformer blocks ---------------------------------------------
    for (int blk = 0; blk < 2; ++blk) {
        const int base = blk * 24;
        // loop order: col, feat, nbr, full  (matches reference)
        const int normI[4] = {base + 19, base + 20, base + 23, base + 22};
        const int wkI[4]   = {base + 0,  base + 4,  base + 12, base + 8};
        const int woI[4]   = {base + 1,  base + 5,  base + 13, base + 9};
        const int wqI[4]   = {base + 2,  base + 6,  base + 14, base + 10};
        const int wvI[4]   = {base + 3,  base + 7,  base + 15, base + 11};
        for (int l = 0; l < 4; ++l) {
            k_rmsnorm<<<M_, 256, 0, stream>>>(x, F(normI[l]), xn);
            gemm(xn, F(wqI[l]), nullptr, qb, M_, DM, DM, 0, stream);
            gemm(xn, F(wkI[l]), nullptr, kb, M_, DM, DM, 0, stream);
            gemm(xn, F(wvI[l]), nullptr, vb, M_, DM, DM, 0, stream);
            k_attn<<<(B_ * NH * S_) / 8, 256, 0, stream>>>(
                qb, kb, vb, node, f2p, colid, tblid, ispad, l, ao);
            gemm(ao, F(woI[l]), nullptr, x, M_, DM, DM, 1, stream); // residual
        }
        // FFN
        k_rmsnorm<<<M_, 256, 0, stream>>>(x, F(base + 21), xn);
        gemm(xn, F(base + 16), nullptr, h1, M_, DFF, DM, 0, stream); // w1
        gemm(xn, F(base + 18), nullptr, h3, M_, DFF, DM, 0, stream); // w3
        k_silu_mul<<<(M_ * DFF) / 256, 256, 0, stream>>>(h1, h3, M_ * DFF);
        gemm(h1, F(base + 17), nullptr, x, M_, DM, DFF, 1, stream);  // w2, residual
    }

    // ---- output norm + loss ---------------------------------------------
    k_rmsnorm<<<M_, 256, 0, stream>>>(x, F(75), xn);
    k_loss<<<M_, 256, 0, stream>>>(xn,
        F(53), F(52),            // dec.number  w,b
        F(51), F(50),            // dec.datetime w,b
        F(49), F(48),            // dec.boolean  w,b
        number_values, datetime_values, boolean_values,
        sem, mks, scal);
    k_final<<<1, 1, 0, stream>>>(scal, (float*)d_out);
}
